// PointNet2Geometric_43035572306073
// MI455X (gfx1250) — matrix operand order repaired and checked by reading, not compiled, not run on hardware
//
#include <hip/hip_runtime.h>

typedef __attribute__((ext_vector_type(16))) _Float16 v16h;
typedef __attribute__((ext_vector_type(8)))  float    v8f;

#define BB   2
#define NPTS 8192

static inline int ceilDiv(int a, int b) { return (a + b - 1) / b; }

// ---------------------------------------------------------------------------
// Kernels
// ---------------------------------------------------------------------------

__global__ void k_scale(const float* __restrict__ x, float* __restrict__ pts, int total) {
  int i = blockIdx.x * blockDim.x + threadIdx.x;
  if (i < total) pts[i] = x[i] * 0.01f;
}

// Farthest point sampling: one block per batch, serial over n samples.
__global__ void k_fps(const float* __restrict__ pos, int Ns, int n,
                      int* __restrict__ outIdx, float* __restrict__ dtmp) {
  int b = blockIdx.x;
  const float* P = pos + (size_t)b * Ns * 3;
  int* I = outIdx + (size_t)b * n;
  float* d = dtmp + (size_t)b * Ns;
  __shared__ float sv[256];
  __shared__ int si[256];
  __shared__ int scur;
  int t = threadIdx.x;
  float x0 = P[0], y0 = P[1], z0 = P[2];
  for (int j = t; j < Ns; j += 256) {
    float dx = P[3*j] - x0, dy = P[3*j+1] - y0, dz = P[3*j+2] - z0;
    d[j] = dx*dx + dy*dy + dz*dz;
  }
  if (t == 0) I[0] = 0;
  __syncthreads();
  for (int i = 1; i < n; i++) {
    float bv = -1.0f; int bi = 0x7fffffff;
    for (int j = t; j < Ns; j += 256) {
      float v = d[j];
      if (v > bv) { bv = v; bi = j; }
    }
    sv[t] = bv; si[t] = bi;
    __syncthreads();
    for (int s = 128; s > 0; s >>= 1) {
      if (t < s) {
        if (sv[t+s] > sv[t] || (sv[t+s] == sv[t] && si[t+s] < si[t])) {
          sv[t] = sv[t+s]; si[t] = si[t+s];
        }
      }
      __syncthreads();
    }
    if (t == 0) { I[i] = si[0]; scur = si[0]; }
    __syncthreads();
    int c = scur;
    float cx = P[3*c], cy = P[3*c+1], cz = P[3*c+2];
    for (int j = t; j < Ns; j += 256) {
      float dx = P[3*j] - cx, dy = P[3*j+1] - cy, dz = P[3*j+2] - cz;
      float v = dx*dx + dy*dy + dz*dz;
      if (v < d[j]) d[j] = v;
    }
    __syncthreads();
  }
}

__global__ void k_gather_pos(const float* __restrict__ pos, const int* __restrict__ idx,
                             int Ns, int Nq, float* __restrict__ pq) {
  int i = blockIdx.x * blockDim.x + threadIdx.x;
  if (i >= BB * Nq) return;
  int b = i / Nq;
  int j = idx[i];
  const float* P = pos + ((size_t)b * Ns + j) * 3;
  float* Q = pq + (size_t)i * 3;
  Q[0] = P[0]; Q[1] = P[1]; Q[2] = P[2];
}

// nearest-32 neighbors (top_k of -d2), valid = d2 <= r2
__global__ void k_ballq(const float* __restrict__ src, const float* __restrict__ q,
                        int Ns, int Nq, float r2,
                        int* __restrict__ nidx, int* __restrict__ nvalid) {
  int id = blockIdx.x * blockDim.x + threadIdx.x;
  if (id >= BB * Nq) return;
  int b = id / Nq;
  const float* P = src + (size_t)b * Ns * 3;
  const float* Q = q + (size_t)id * 3;
  float qx = Q[0], qy = Q[1], qz = Q[2];
  float bd[32]; int bi[32];
  for (int k = 0; k < 32; k++) { bd[k] = 3.4e38f; bi[k] = 0; }
  for (int j = 0; j < Ns; j++) {
    float dx = P[3*j] - qx, dy = P[3*j+1] - qy, dz = P[3*j+2] - qz;
    float d2 = dx*dx + dy*dy + dz*dz;
    if (d2 < bd[31]) {
      int k = 31;
      while (k > 0 && bd[k-1] > d2) { bd[k] = bd[k-1]; bi[k] = bi[k-1]; k--; }
      bd[k] = d2; bi[k] = j;
    }
  }
  int* O = nidx + (size_t)id * 32;
  int* V = nvalid + (size_t)id * 32;
  for (int k = 0; k < 32; k++) { O[k] = bi[k]; V[k] = (bd[k] <= r2) ? 1 : 0; }
}

// build SA edge features: [x[nbr] (Cx), rel (3), zero-pad] -> f16 rows of width Kpad
__global__ void k_edges(const float* __restrict__ src, const float* __restrict__ q,
                        const float* __restrict__ xs, int Cx, int Ns, int Nq, int Kpad,
                        const int* __restrict__ nidx, _Float16* __restrict__ A) {
  int id = blockIdx.x * blockDim.x + threadIdx.x;
  int total = BB * Nq * 32;
  if (id >= total) return;
  int bq = id >> 5;
  int b = bq / Nq;
  int j = nidx[id];
  const float* P = src + ((size_t)b * Ns + j) * 3;
  const float* Q = q + (size_t)bq * 3;
  _Float16* a = A + (size_t)id * Kpad;
  int c = 0;
  if (xs) {
    const float* X = xs + ((size_t)b * Ns + j) * Cx;
    for (; c < Cx; c++) a[c] = (_Float16)X[c];
  }
  a[c++] = (_Float16)(P[0] - Q[0]);
  a[c++] = (_Float16)(P[1] - Q[1]);
  a[c++] = (_Float16)(P[2] - Q[2]);
  for (; c < Kpad; c++) a[c] = (_Float16)0.0f;
}

// weight f32 (K x C) -> f16 packed directly in WMMA B-fragment layout:
// Wp[((kb*tilesN + nt)*32 + lane)*16 + i] = W[kb*32 + (lane>>4)*16 + i, nt*16 + (lane&15)]
// so each lane's 16 halfs for a (K-block, N-tile) are 32 contiguous bytes, and
// adjacent N-tiles for the same K-block are 1024 bytes apart.
__global__ void k_wpack(const float* __restrict__ W, int K, int C, int Kpad,
                        _Float16* __restrict__ Wp) {
  int id = blockIdx.x * blockDim.x + threadIdx.x;
  if (id >= Kpad * C) return;
  int i    = id & 15;
  int lane = (id >> 4) & 31;
  int rest = id >> 9;           // kb*tilesN + nt
  int tilesN = C >> 4;
  int kb = rest / tilesN;
  int nt = rest % tilesN;
  int k = (kb << 5) + ((lane >> 4) << 4) + i;
  int c = (nt << 4) + (lane & 15);
  Wp[id] = (k < K) ? (_Float16)W[(size_t)k * C + c] : (_Float16)0.0f;
}

// H = ReLU(A @ W + bias); A: M x Kpad f16 row-major, Wp: packed B fragments.
// One wave computes NT adjacent 16x16 N-tiles for one M-tile, reusing the A
// fragment across NT v_wmma_f32_16x16x32_f16 issues per K-step. All fragment
// loads are immediate-offset b128s from two pointers bumped once per K-step.
template <int NT>
__global__ void k_gemm(const _Float16* __restrict__ A, const _Float16* __restrict__ Wp,
                       const float* __restrict__ bias, float* __restrict__ Hout,
                       int M, int Kpad, int C) {
  int tilesN  = C >> 4;
  int groupsN = tilesN / NT;
  int wave = (blockIdx.x * blockDim.x + threadIdx.x) >> 5;
  int lane = threadIdx.x & 31;
  int groups = (M >> 4) * groupsN;
  if (wave >= groups) return;
  int tm  = (wave / groupsN) << 4;
  int ng0 = (wave % groupsN) * NT;
  int row   = tm + (lane & 15);
  int koffA = (lane >> 4) << 4;   // 0 or 16
  v8f zero = {};
  v8f acc[NT];
#pragma unroll
  for (int t = 0; t < NT; t++) acc[t] = zero;
  const _Float16* Ar = A + (size_t)row * Kpad + koffA;        // bump by 32 halfs / K-step
  const _Float16* Bp = Wp + ((size_t)ng0 * 32 + lane) * 16;   // bump by tilesN*512 halfs
  size_t bstep = (size_t)tilesN * 512;
  int nKB = Kpad >> 5;
  for (int kb = 0; kb < nKB; kb++) {
    v16h av;
#pragma unroll
    for (int i = 0; i < 16; i++) { av[i] = Ar[i]; }
#pragma unroll
    for (int t = 0; t < NT; t++) {
      const _Float16* bp = Bp + (size_t)t * 512;   // immediate offsets t*1024 bytes
      v16h bv;
#pragma unroll
      for (int i = 0; i < 16; i++) bv[i] = bp[i];
      acc[t] = __builtin_amdgcn_wmma_f32_16x16x32_f16(false, av, false, bv,
                                                      (short)0, acc[t], false, false);
    }
    Ar += 32;
    Bp += bstep;
  }
  int r0 = tm + ((lane >> 4) << 3);
#pragma unroll
  for (int t = 0; t < NT; t++) {
    int col = ((ng0 + t) << 4) + (lane & 15);
    float bc = bias[col];
#pragma unroll
    for (int r = 0; r < 8; r++) {
      float v = acc[t][r] + bc;
      v = v > 0.0f ? v : 0.0f;
      Hout[(size_t)(r0 + r) * C + col] = v;
    }
  }
}

__global__ void k_zero(float* __restrict__ p, int n) {
  int i = blockIdx.x * blockDim.x + threadIdx.x;
  if (i < n) p[i] = 0.0f;
}

__global__ void k_count(const int* __restrict__ mask, int rows, float* __restrict__ cnt) {
  __shared__ float sh[256];
  float loc = 0.0f;
  for (int i = blockIdx.x * blockDim.x + threadIdx.x; i < rows; i += gridDim.x * blockDim.x)
    loc += (mask[i] != 0) ? 1.0f : 0.0f;
  sh[threadIdx.x] = loc;
  __syncthreads();
  for (int s = 128; s > 0; s >>= 1) {
    if (threadIdx.x < s) sh[threadIdx.x] += sh[threadIdx.x + s];
    __syncthreads();
  }
  if (threadIdx.x == 0) atomicAdd(cnt, sh[0]);
}

// masked per-channel sum / sumsq
__global__ void k_stats(const float* __restrict__ H, const int* __restrict__ mask,
                        int rows, int C, float* __restrict__ sum, float* __restrict__ ssq) {
  extern __shared__ float sh[];  // 2*C floats
  for (int c = threadIdx.x; c < 2 * C; c += blockDim.x) sh[c] = 0.0f;
  __syncthreads();
  int total = rows * C;
  for (int i = blockIdx.x * blockDim.x + threadIdx.x; i < total; i += gridDim.x * blockDim.x) {
    int r = i / C, c = i % C;
    if (!mask || mask[r]) {
      float v = H[i];
      atomicAdd(&sh[c], v);
      atomicAdd(&sh[C + c], v * v);
    }
  }
  __syncthreads();
  for (int c = threadIdx.x; c < C; c += blockDim.x) {
    atomicAdd(&sum[c], sh[c]);
    atomicAdd(&ssq[c], sh[C + c]);
  }
}

__global__ void k_bnapply(float* __restrict__ H, _Float16* __restrict__ Anext,
                          const float* __restrict__ sum, const float* __restrict__ ssq,
                          const float* __restrict__ cntPtr, float cntConst,
                          const float* __restrict__ gamma, const float* __restrict__ beta,
                          int rows, int C, int writeHalf) {
  int i = blockIdx.x * blockDim.x + threadIdx.x;
  if (i >= rows * C) return;
  int c = i % C;
  float cnt = cntPtr ? cntPtr[0] : cntConst;
  cnt = fmaxf(cnt, 1.0f);
  float mean = sum[c] / cnt;
  float var = ssq[c] / cnt - mean * mean;
  if (var < 0.0f) var = 0.0f;
  float v = gamma[c] * (H[i] - mean) * rsqrtf(var + 1e-5f) + beta[c];
  H[i] = v;
  if (writeHalf) Anext[i] = (_Float16)v;  // next layer's Kpad == C (multiple of 32)
}

__global__ void k_maxaggr(const float* __restrict__ H, const int* __restrict__ valid,
                          int Nq, int C, float* __restrict__ xout) {
  int id = blockIdx.x * blockDim.x + threadIdx.x;
  if (id >= BB * Nq * C) return;
  int c = id % C;
  int bq = id / C;
  const float* hrow = H + ((size_t)bq * 32) * C + c;
  const int* vrow = valid + (size_t)bq * 32;
  float m = -INFINITY;
  for (int k = 0; k < 32; k++)
    if (vrow[k]) { float v = hrow[(size_t)k * C]; if (v > m) m = v; }
  xout[id] = m;
}

__global__ void k_knn3(const float* __restrict__ src, int Ns,
                       const float* __restrict__ tgt, int Nt,
                       int* __restrict__ ki, float* __restrict__ kw) {
  int id = blockIdx.x * blockDim.x + threadIdx.x;
  if (id >= BB * Nt) return;
  int b = id / Nt;
  const float* P = src + (size_t)b * Ns * 3;
  const float* Q = tgt + (size_t)id * 3;
  float qx = Q[0], qy = Q[1], qz = Q[2];
  float bd0 = 3.4e38f, bd1 = 3.4e38f, bd2 = 3.4e38f;
  int bi0 = 0, bi1 = 0, bi2 = 0;
  for (int j = 0; j < Ns; j++) {
    float dx = P[3*j] - qx, dy = P[3*j+1] - qy, dz = P[3*j+2] - qz;
    float d = dx*dx + dy*dy + dz*dz;
    if (d < bd2) {
      if (d < bd0)      { bd2 = bd1; bi2 = bi1; bd1 = bd0; bi1 = bi0; bd0 = d; bi0 = j; }
      else if (d < bd1) { bd2 = bd1; bi2 = bi1; bd1 = d; bi1 = j; }
      else              { bd2 = d; bi2 = j; }
    }
  }
  float w0 = 1.0f / fmaxf(bd0, 1e-16f);
  float w1 = 1.0f / fmaxf(bd1, 1e-16f);
  float w2 = 1.0f / fmaxf(bd2, 1e-16f);
  int* I = ki + (size_t)id * 3;
  I[0] = bi0; I[1] = bi1; I[2] = bi2;
  float* W = kw + (size_t)id * 4;
  W[0] = w0; W[1] = w1; W[2] = w2; W[3] = w0 + w1 + w2;
}

__global__ void k_interp(const float* __restrict__ xs, int C,
                         const int* __restrict__ ki, const float* __restrict__ kw,
                         int Nt, int Ns, float* __restrict__ out) {
  int id = blockIdx.x * blockDim.x + threadIdx.x;
  if (id >= BB * Nt * C) return;
  int c = id % C;
  int bt = id / C;
  int b = bt / Nt;
  const int* I = ki + (size_t)bt * 3;
  const float* W = kw + (size_t)bt * 4;
  const float* X = xs + (size_t)b * Ns * C;
  out[id] = (W[0] * X[(size_t)I[0] * C + c] +
             W[1] * X[(size_t)I[1] * C + c] +
             W[2] * X[(size_t)I[2] * C + c]) / W[3];
}

__global__ void k_fpedges(const float* __restrict__ interp, int Cint,
                          const float* __restrict__ skip, int Cskip, int Nt,
                          _Float16* __restrict__ A) {
  int Wd = Cint + Cskip;  // always a multiple of 32
  int id = blockIdx.x * blockDim.x + threadIdx.x;
  if (id >= BB * Nt * Wd) return;
  int c = id % Wd;
  int r = id / Wd;
  float v = (c < Cint) ? interp[(size_t)r * Cint + c]
                       : skip[(size_t)r * Cskip + (c - Cint)];
  A[id] = (_Float16)v;
}

__global__ void k_copyf(const float* __restrict__ src, float* __restrict__ dst, int n) {
  int i = blockIdx.x * blockDim.x + threadIdx.x;
  if (i < n) dst[i] = src[i];
}

__global__ void k_transout(const float* __restrict__ y, float* __restrict__ out) {
  int id = blockIdx.x * blockDim.x + threadIdx.x;
  if (id >= BB * NPTS * 128) return;
  int c = id % 128;
  int bt = id / 128;
  int b = bt / NPTS;
  int t = bt % NPTS;
  out[((size_t)b * 128 + c) * NPTS + t] = y[id];
}

// ---------------------------------------------------------------------------
// Host orchestration
// ---------------------------------------------------------------------------

// layer specs, internal module order: sa1,sa2,sa3,sa4,fp4,fp3,fp2,fp1
static const int LK_[21] = {3,32,32, 67,64,64, 131,128,128, 259,256,256,
                            768,256, 384,256, 320,256, 128,128,128};
static const int LC_[21] = {32,32,64, 64,64,128, 128,128,256, 256,256,512,
                            256,256, 256,256, 256,128, 128,128,128};
static const int MODBASE_[8] = {0,3,6,9,12,14,16,18};
static const int MODNL_[8]   = {3,3,3,3,2,2,2,3};

struct Params {
  const float *W[21], *bias[21], *gamma[21], *beta[21];
  const float *x;
};

static void resolve(void* const* d_in, const int* in_sizes, Params& P) {
  bool xFirst = (in_sizes[0] == BB * NPTS * 3);  // python insertion order vs jax sorted order
  int pos = 0;
  if (xFirst) {
    // insertion order: x, sa1..sa4, fp4, fp3, fp2, fp1; layer dict: W, b, gamma, beta
    P.x = (const float*)d_in[pos++];
    static const int order[8] = {0,1,2,3,4,5,6,7};
    for (int m = 0; m < 8; m++)
      for (int l = 0; l < MODNL_[order[m]]; l++) {
        int gi = MODBASE_[order[m]] + l;
        P.W[gi]     = (const float*)d_in[pos++];
        P.bias[gi]  = (const float*)d_in[pos++];
        P.gamma[gi] = (const float*)d_in[pos++];
        P.beta[gi]  = (const float*)d_in[pos++];
      }
  } else {
    // jax pytree sorted: fp1,fp2,fp3,fp4,sa1..sa4 then x; layer dict: W, b, beta, gamma
    static const int order[8] = {7,6,5,4,0,1,2,3};
    for (int m = 0; m < 8; m++)
      for (int l = 0; l < MODNL_[order[m]]; l++) {
        int gi = MODBASE_[order[m]] + l;
        P.W[gi]     = (const float*)d_in[pos++];
        P.bias[gi]  = (const float*)d_in[pos++];
        P.beta[gi]  = (const float*)d_in[pos++];
        P.gamma[gi] = (const float*)d_in[pos++];
      }
    P.x = (const float*)d_in[pos++];
  }
}

static void run_mlp(int base, int nl, int rows, const int* rowmask, const Params& P,
                    _Float16* Ah, _Float16* Wh, float* Hb, float* stats, hipStream_t s) {
  for (int l = 0; l < nl; l++) {
    int gi = base + l;
    int K = LK_[gi], C = LC_[gi];
    int Kpad = (K + 31) & ~31;
    { int tot = Kpad * C;
      k_wpack<<<ceilDiv(tot, 256), 256, 0, s>>>(P.W[gi], K, C, Kpad, Wh); }
    { int tilesN = C / 16;
      if (tilesN % 4 == 0) {
        int groups = (rows / 16) * (tilesN / 4);
        k_gemm<4><<<ceilDiv(groups, 8), 256, 0, s>>>(Ah, Wh, P.bias[gi], Hb, rows, Kpad, C);
      } else {
        int groups = (rows / 16) * (tilesN / 2);
        k_gemm<2><<<ceilDiv(groups, 8), 256, 0, s>>>(Ah, Wh, P.bias[gi], Hb, rows, Kpad, C);
      }
    }
    k_zero<<<ceilDiv(1025, 256), 256, 0, s>>>(stats, 1025);
    float* sum = stats;
    float* ssq = stats + 512;
    float* cnt = stats + 1024;
    if (rowmask) k_count<<<64, 256, 0, s>>>(rowmask, rows, cnt);
    { int tot = rows * C;
      int blocks = ceilDiv(tot, 256);
      if (blocks > 1024) blocks = 1024;
      k_stats<<<blocks, 256, 2 * C * (int)sizeof(float), s>>>(Hb, rowmask, rows, C, sum, ssq); }
    { int tot = rows * C;
      int wh = (l < nl - 1) ? 1 : 0;
      k_bnapply<<<ceilDiv(tot, 256), 256, 0, s>>>(Hb, Ah, sum, ssq,
                                                  rowmask ? cnt : nullptr, (float)rows,
                                                  P.gamma[gi], P.beta[gi], rows, C, wh); }
  }
}

extern "C" void kernel_launch(void* const* d_in, const int* in_sizes, int n_in,
                              void* d_out, int out_size, void* d_ws, size_t ws_size,
                              hipStream_t stream) {
  (void)out_size;
  if (n_in < 85) return;
  Params P;
  resolve(d_in, in_sizes, P);

  const int NSv[4]   = {NPTS, 4096, 1024, 256};
  const int NQv[4]   = {4096, 1024, 256, 64};
  const float R2v[4] = {0.0025f, 0.01f, 0.04f, 0.16f};
  const int CXv[4]   = {0, 64, 128, 256};
  const int COUTv[4] = {64, 128, 256, 512};

  // bump allocator over d_ws
  char* base = (char*)d_ws;
  size_t off = 0;
  auto take = [&](size_t bytes) -> void* {
    size_t o = (off + 255) & ~(size_t)255;
    off = o + bytes;
    return (void*)(base + o);
  };

  float* pts = (float*)take((size_t)BB * NPTS * 3 * 4);
  float* pq[4];
  for (int l = 0; l < 4; l++) pq[l] = (float*)take((size_t)BB * NQv[l] * 3 * 4);
  int* fpsidx = (int*)take((size_t)BB * 4096 * 4);
  float* fpsd = (float*)take((size_t)BB * NPTS * 4);
  int* nidx   = (int*)take((size_t)BB * 4096 * 32 * 4);
  int* nvalid = (int*)take((size_t)BB * 4096 * 32 * 4);
  float* xsa[4];
  for (int l = 0; l < 4; l++) xsa[l] = (float*)take((size_t)BB * NQv[l] * COUTv[l] * 4);
  float* yfp4 = (float*)take((size_t)BB * 256 * 256 * 4);
  float* yfp3 = (float*)take((size_t)BB * 1024 * 256 * 4);
  float* yfp2 = (float*)take((size_t)BB * 4096 * 128 * 4);
  float* interp = (float*)take((size_t)2097152 * 4);
  int* knni   = (int*)take((size_t)BB * NPTS * 3 * 4);
  float* knnw = (float*)take((size_t)BB * NPTS * 4 * 4);
  float* stats = (float*)take((size_t)1025 * 4);
  _Float16* Ah = (_Float16*)take((size_t)8388608 * 2);
  _Float16* Wh = (_Float16*)take((size_t)196608 * 2);
  float* Hb = (float*)take((size_t)16777216 * 4);
  if (off > ws_size) return;

  k_scale<<<ceilDiv(BB * NPTS * 3, 256), 256, 0, stream>>>(P.x, pts, BB * NPTS * 3);

  // -------- SA modules (with interleaved FPS sampling chain) --------
  const float* srcPos = pts;
  for (int l = 0; l < 4; l++) {
    int Ns = NSv[l], Nq = NQv[l];
    k_fps<<<BB, 256, 0, stream>>>(srcPos, Ns, Nq, fpsidx, fpsd);
    k_gather_pos<<<ceilDiv(BB * Nq, 256), 256, 0, stream>>>(srcPos, fpsidx, Ns, Nq, pq[l]);
    k_ballq<<<ceilDiv(BB * Nq, 128), 128, 0, stream>>>(srcPos, pq[l], Ns, Nq, R2v[l],
                                                       nidx, nvalid);
    int gi0 = MODBASE_[l];
    int Kpad0 = (LK_[gi0] + 31) & ~31;
    const float* xsrc = (l == 0) ? nullptr : xsa[l - 1];
    k_edges<<<ceilDiv(BB * Nq * 32, 128), 128, 0, stream>>>(srcPos, pq[l], xsrc, CXv[l],
                                                            Ns, Nq, Kpad0, nidx, Ah);
    int rows = BB * Nq * 32;
    run_mlp(gi0, 3, rows, nvalid, P, Ah, Wh, Hb, stats, stream);
    k_maxaggr<<<ceilDiv(BB * Nq * COUTv[l], 256), 256, 0, stream>>>(Hb, nvalid, Nq,
                                                                    COUTv[l], xsa[l]);
    srcPos = pq[l];
  }

  // -------- FP modules --------
  // fp4: x4@p4(64,512) -> p3(256), skip x3(256)
  k_knn3<<<ceilDiv(BB * 256, 128), 128, 0, stream>>>(pq[3], 64, pq[2], 256, knni, knnw);
  k_interp<<<ceilDiv(BB * 256 * 512, 256), 256, 0, stream>>>(xsa[3], 512, knni, knnw,
                                                             256, 64, interp);
  k_fpedges<<<ceilDiv(BB * 256 * 768, 256), 256, 0, stream>>>(interp, 512, xsa[2], 256,
                                                              256, Ah);
  run_mlp(12, 2, BB * 256, nullptr, P, Ah, Wh, Hb, stats, stream);
  k_copyf<<<ceilDiv(BB * 256 * 256, 256), 256, 0, stream>>>(Hb, yfp4, BB * 256 * 256);

  // fp3: yfp4@p3(256,256) -> p2(1024), skip x2(128)
  k_knn3<<<ceilDiv(BB * 1024, 128), 128, 0, stream>>>(pq[2], 256, pq[1], 1024, knni, knnw);
  k_interp<<<ceilDiv(BB * 1024 * 256, 256), 256, 0, stream>>>(yfp4, 256, knni, knnw,
                                                              1024, 256, interp);
  k_fpedges<<<ceilDiv(BB * 1024 * 384, 256), 256, 0, stream>>>(interp, 256, xsa[1], 128,
                                                               1024, Ah);
  run_mlp(14, 2, BB * 1024, nullptr, P, Ah, Wh, Hb, stats, stream);
  k_copyf<<<ceilDiv(BB * 1024 * 256, 256), 256, 0, stream>>>(Hb, yfp3, BB * 1024 * 256);

  // fp2: yfp3@p2(1024,256) -> p1(4096), skip x1(64)
  k_knn3<<<ceilDiv(BB * 4096, 128), 128, 0, stream>>>(pq[1], 1024, pq[0], 4096, knni, knnw);
  k_interp<<<ceilDiv(BB * 4096 * 256, 256), 256, 0, stream>>>(yfp3, 256, knni, knnw,
                                                              4096, 1024, interp);
  k_fpedges<<<ceilDiv(BB * 4096 * 320, 256), 256, 0, stream>>>(interp, 256, xsa[0], 64,
                                                               4096, Ah);
  run_mlp(16, 2, BB * 4096, nullptr, P, Ah, Wh, Hb, stats, stream);
  k_copyf<<<ceilDiv(BB * 4096 * 128, 256), 256, 0, stream>>>(Hb, yfp2, BB * 4096 * 128);

  // fp1: yfp2@p1(4096,128) -> pts(8192), no skip
  k_knn3<<<ceilDiv(BB * NPTS, 128), 128, 0, stream>>>(pq[0], 4096, pts, NPTS, knni, knnw);
  k_interp<<<ceilDiv(BB * NPTS * 128, 256), 256, 0, stream>>>(yfp2, 128, knni, knnw,
                                                              NPTS, 4096, interp);
  k_fpedges<<<ceilDiv(BB * NPTS * 128, 256), 256, 0, stream>>>(interp, 128, nullptr, 0,
                                                               NPTS, Ah);
  run_mlp(18, 3, BB * NPTS, nullptr, P, Ah, Wh, Hb, stats, stream);

  k_transout<<<ceilDiv(BB * NPTS * 128, 256), 256, 0, stream>>>(Hb, (float*)d_out);
}
